// PAMModule_87316685127756
// MI455X (gfx1250) — matrix operand order repaired and checked by reading, not compiled, not run on hardware
//
#include <hip/hip_runtime.h>

// ---------------------------------------------------------------------------
// PAM attention for MI455X (gfx1250): bf16 WMMA, flash-attention fusion,
// double-buffered async global->LDS staging for K tiles.
//   x: [4,512,64,64] f32 ; Wq/Wk: [64,512] ; Wv: [512,512] ; biases ; gamma
//   out = gamma * (softmax(Q K^T) V^T)^T + x
// Workspace: qt bf16 [4][4096][64], kt bf16 [4][4096][64], vt bf16 [4][512][4096]
// ---------------------------------------------------------------------------

typedef __bf16 bf16_t;
typedef __attribute__((ext_vector_type(16))) __bf16 v16bf;
typedef __attribute__((ext_vector_type(8)))  __bf16 v8bf;
typedef __attribute__((ext_vector_type(8)))  float  v8f;

#define WMMA_BF16(A, B, C) \
  __builtin_amdgcn_wmma_f32_16x16x32_bf16(false, (A), false, (B), (short)0, (C), false, false)

#define N_PIX 4096
#define CHN   512

__device__ __forceinline__ unsigned pack_bf16(float a, float b) {
  union { bf16_t h[2]; unsigned u; } p;
  p.h[0] = (bf16_t)a; p.h[1] = (bf16_t)b;
  return p.u;                                    // -> v_cvt_pk_bf16_f32
}

// ---- fragment loaders (per CDNA5 ISA §7.12.2 16-bit layouts) ----------------
// 32-bit offset math keeps addressing on saddr + voffset forms.
// A (16x32, MxK), row-major [M][K] storage with leading dim ld (elements).
__device__ __forceinline__ v16bf load_frag_a(const bf16_t* base, int row, int ld, int lane) {
  const int m  = lane & 15;
  const int hk = lane >> 4;
  const unsigned o = (unsigned)((row + m) * ld + hk * 8);
  const v8bf* p0 = (const v8bf*)(base + o);
  const v8bf* p1 = (const v8bf*)(base + o + 16);
  v8bf lo = *p0, hi = *p1;
  v16bf r;
#pragma unroll
  for (int i = 0; i < 8; ++i) { r[i] = lo[i]; r[i + 8] = hi[i]; }
  return r;
}

// B (32x16, KxN) loaded from TRANSPOSED storage Bt[N][K] (leading dim ld).
__device__ __forceinline__ v16bf load_frag_b(const bf16_t* baseT, int ld, int lane) {
  const int n  = lane & 15;
  const int hk = lane >> 4;
  const unsigned o = (unsigned)(n * ld + hk * 8);
  const v8bf* p0 = (const v8bf*)(baseT + o);
  const v8bf* p1 = (const v8bf*)(baseT + o + 16);
  v8bf lo = *p0, hi = *p1;
  v16bf r;
#pragma unroll
  for (int i = 0; i < 8; ++i) { r[i] = lo[i]; r[i + 8] = hi[i]; }
  return r;
}

// ---------------------------------------------------------------------------
// Projection: out[pix][o] = sum_k x[k][pix] * W[o][k] + b[o]   (per batch)
// blockIdx: x = pixel tile (128), y = output tile 0..9 (0=q,1=k,2..9=v), z = batch
// ---------------------------------------------------------------------------
__global__ __launch_bounds__(256) void proj_kernel(
    const float* __restrict__ x,
    const float* __restrict__ Wq, const float* __restrict__ bq,
    const float* __restrict__ Wk, const float* __restrict__ bk,
    const float* __restrict__ Wv, const float* __restrict__ bv,
    bf16_t* __restrict__ qt, bf16_t* __restrict__ kt, bf16_t* __restrict__ vt)
{
  const int LDK = 40;                       // padded leading dim (80B, 16B aligned)
  __shared__ bf16_t As[128 * LDK];          // x tile, transposed: [pix][k]
  __shared__ bf16_t Bs[64 * LDK];           // weight tile: [out][k]

  const int b    = blockIdx.z;
  const int pix0 = blockIdx.x * 128;
  const int y    = blockIdx.y;
  const int t    = threadIdx.x;
  const int lane = t & 31;
  const int w    = t >> 5;
  const int wr   = w & 3;                   // 32-pixel group (0..3)
  const int wc   = w >> 2;                  // 32-output group (0..1)

  const float* Wsrc;
  const float* bsrc;
  if (y == 0)      { Wsrc = Wq;                      bsrc = bq; }
  else if (y == 1) { Wsrc = Wk;                      bsrc = bk; }
  else             { Wsrc = Wv + (y - 2) * 64 * CHN; bsrc = bv + (y - 2) * 64; }

  // destination base + 32-bit strides (hoists q/k/v branch out of store loop)
  bf16_t* dst;
  int sp, so;                               // element strides for pixel / out-local
  if (y == 0)      { dst = qt + (size_t)b * N_PIX * 64 + (size_t)pix0 * 64; sp = 64; so = 1; }
  else if (y == 1) { dst = kt + (size_t)b * N_PIX * 64 + (size_t)pix0 * 64; sp = 64; so = 1; }
  else             { dst = vt + (size_t)b * CHN * N_PIX + (size_t)(y - 2) * 64 * N_PIX + pix0;
                     sp = 1; so = N_PIX; }

  const float* xb = x + (size_t)b * CHN * N_PIX;

  v8f acc[2][2];
#pragma unroll
  for (int i = 0; i < 2; ++i)
#pragma unroll
    for (int j = 0; j < 2; ++j)
#pragma unroll
      for (int r = 0; r < 8; ++r) acc[i][j][r] = 0.f;

  for (int k0 = 0; k0 < CHN; k0 += 32) {
    // stage x tile [32 k x 128 pix] -> bf16 [pix][k]; float2 loads, packed b32 stores
#pragma unroll
    for (int i = 0; i < 4; ++i) {
      int ci = t + i * 256;                  // 0..1023 groups of 2k x 2pix
      int k  = (ci >> 6) * 2;                // 0,2,..,30
      int p  = (ci & 63) * 2;                // 0,2,..,126
      float2 f0 = *(const float2*)(xb + (unsigned)((k0 + k)     * N_PIX + pix0 + p));
      float2 f1 = *(const float2*)(xb + (unsigned)((k0 + k + 1) * N_PIX + pix0 + p));
      *(unsigned*)&As[(p)     * LDK + k] = pack_bf16(f0.x, f1.x);
      *(unsigned*)&As[(p + 1) * LDK + k] = pack_bf16(f0.y, f1.y);
    }
    // stage weight tile [64 out x 32 k]
#pragma unroll
    for (int i = 0; i < 4; ++i) {
      int ci = t + i * 256;                  // 0..1023 pairs
      int oo = ci >> 4;                      // 0..63
      int k  = (ci & 15) * 2;                // 0,2,..,30
      float2 wv = *(const float2*)(Wsrc + (unsigned)(oo * CHN + k0 + k));
      *(unsigned*)&Bs[oo * LDK + k] = pack_bf16(wv.x, wv.y);
    }
    __syncthreads();

    v16bf a0 = load_frag_a(As, wr * 32,      LDK, lane);
    v16bf a1 = load_frag_a(As, wr * 32 + 16, LDK, lane);
#pragma unroll
    for (int tj = 0; tj < 2; ++tj) {
      v16bf bb = load_frag_b(Bs + (wc * 32 + tj * 16) * LDK, LDK, lane);
      acc[0][tj] = WMMA_BF16(a0, bb, acc[0][tj]);
      acc[1][tj] = WMMA_BF16(a1, bb, acc[1][tj]);
    }
    __syncthreads();
  }

  // epilogue: bias add, bf16 convert, branch-free strided store (32-bit offsets)
  const int hk = lane >> 4, nn = lane & 15;
#pragma unroll
  for (int ti = 0; ti < 2; ++ti)
#pragma unroll
    for (int tj = 0; tj < 2; ++tj)
#pragma unroll
      for (int r = 0; r < 8; ++r) {
        int pl = wr * 32 + ti * 16 + r + hk * 8;      // local pixel
        int ol = wc * 32 + tj * 16 + nn;              // local output
        float v = acc[ti][tj][r] + bsrc[ol];
        dst[(unsigned)(pl * sp + ol * so)] = (bf16_t)v;
      }
}

// ---------------------------------------------------------------------------
// Async stage of one K tile (64 rows x 64 red = 8KB) into padded LDS buffer.
// Per-lane LDS destinations; 2 x GLOBAL_LOAD_ASYNC_TO_LDS_B128 per wave.
// ---------------------------------------------------------------------------
#define LDKK 72
__device__ __forceinline__ void stage_k_async(const bf16_t* kb, int kt0,
                                              bf16_t* ksh, int t) {
#pragma unroll
  for (int j = 0; j < 2; ++j) {
    int ci    = t + j * 256;                 // 0..511 (row, 16B chunk)
    int row   = ci >> 3;
    int chunk = ci & 7;
    unsigned lds_addr = (unsigned)(size_t)&ksh[row * LDKK + chunk * 8];
    unsigned long long ga =
        (unsigned long long)(size_t)(kb + (size_t)(kt0 + row) * 64 + chunk * 8);
    asm volatile("global_load_async_to_lds_b128 %0, %1, off"
                 :: "v"(lds_addr), "v"(ga)
                 : "memory");
  }
}

// ---------------------------------------------------------------------------
// Fused flash attention + residual.
// block = 64 query rows x full 512 channels ; 8 waves:
//   energy: wave w owns S tiles (rt=w>>1, ct={2(w&1), 2(w&1)+1})
//   output: wave w owns channels [64w, 64w+64)
// ---------------------------------------------------------------------------
__global__ __launch_bounds__(256) void attn_kernel(
    const float* __restrict__ x, const float* __restrict__ gamma_p,
    const bf16_t* __restrict__ qt, const bf16_t* __restrict__ kt,
    const bf16_t* __restrict__ vt, float* __restrict__ out)
{
  const int LDS_S = 65;                      // f32 leading dim (pad)
  const int LDP   = 72;                      // bf16 leading dim (144B, 16B aligned)
  __shared__ float  Ssh[64 * LDS_S];
  __shared__ bf16_t Psh[64 * LDP];
  __shared__ bf16_t Ksh[2][64 * LDKK];       // double-buffered K tile
  __shared__ float  mrow[64], lrow[64], arow[64];

  const int b    = blockIdx.y;
  const int q0   = blockIdx.x * 64;
  const int t    = threadIdx.x;
  const int lane = t & 31;
  const int w    = t >> 5;
  const int hk   = lane >> 4, nn = lane & 15;

  const float gamma = gamma_p[0];

  const bf16_t* qb = qt + (size_t)b * N_PIX * 64;
  const bf16_t* kb = kt + (size_t)b * N_PIX * 64;
  const bf16_t* vb = vt + (size_t)b * CHN * N_PIX;

  const int rt  = w >> 1;                    // energy row tile (0..3)
  const int ct0 = (w & 1) * 2;               // energy col tiles ct0, ct0+1
  const int ch0 = w * 64;                    // output channel group

  // Q fragments for this wave's energy row tile (reused every key tile)
  v16bf qf[2];
#pragma unroll
  for (int ks = 0; ks < 2; ++ks)
    qf[ks] = load_frag_a(qb + (size_t)(q0 + rt * 16) * 64 + ks * 32, 0, 64, lane);

  v8f acc[4][4];
#pragma unroll
  for (int i = 0; i < 4; ++i)
#pragma unroll
    for (int j = 0; j < 4; ++j)
#pragma unroll
      for (int r = 0; r < 8; ++r) acc[i][j][r] = 0.f;

  if (t < 64) { mrow[t] = -1e30f; lrow[t] = 0.f; }

  // prologue: kick off async copy of first K tile
  stage_k_async(kb, 0, Ksh[0], t);
  int cur = 0;
  __syncthreads();

  for (int kt0 = 0; kt0 < N_PIX; kt0 += 64) {
    // ---- 0) prefetch next K tile into the other buffer; wait current ----
    if (kt0 + 64 < N_PIX) {
      stage_k_async(kb, kt0 + 64, Ksh[cur ^ 1], t);
      asm volatile("s_wait_asynccnt 0x2" ::: "memory");   // current tile done
    } else {
      asm volatile("s_wait_asynccnt 0x0" ::: "memory");
    }
    __syncthreads();
    const bf16_t* kcur = Ksh[cur];

    // ---- 1) S = Q K^T for this 64x64 tile -------------------------------
#pragma unroll
    for (int c = 0; c < 2; ++c) {
      int ct = ct0 + c;
      v8f s;
#pragma unroll
      for (int r = 0; r < 8; ++r) s[r] = 0.f;
#pragma unroll
      for (int ks = 0; ks < 2; ++ks) {
        v16bf bf = load_frag_b(kcur + (ct * 16) * LDKK + ks * 32, LDKK, lane);
        s = WMMA_BF16(qf[ks], bf, s);
      }
#pragma unroll
      for (int r = 0; r < 8; ++r)
        Ssh[(rt * 16 + r + hk * 8) * LDS_S + ct * 16 + nn] = s[r];
    }
    __syncthreads();

    // ---- 2) online softmax: 4 threads per row, quad shuffles ------------
    {
      int row = t >> 2, qd = t & 3;
      const float* srow = &Ssh[row * LDS_S + qd * 16];
      float mo = mrow[row];
      float mt = mo;
#pragma unroll
      for (int j = 0; j < 16; ++j) mt = fmaxf(mt, srow[j]);
      mt = fmaxf(mt, __shfl_xor(mt, 1, 32));
      mt = fmaxf(mt, __shfl_xor(mt, 2, 32));
      float ls = 0.f;
#pragma unroll
      for (int j = 0; j < 16; ++j) {
        float p = __expf(srow[j] - mt);
        Psh[row * LDP + qd * 16 + j] = (bf16_t)p;
        ls += p;
      }
      ls += __shfl_xor(ls, 1, 32);
      ls += __shfl_xor(ls, 2, 32);
      if (qd == 0) {
        float al = __expf(mo - mt);          // mo=-1e30 initially -> 0
        mrow[row] = mt;
        lrow[row] = lrow[row] * al + ls;
        arow[row] = al;
      }
    }
    __syncthreads();

    // ---- 3) rescale accumulators, O += P V^T ----------------------------
#pragma unroll
    for (int it = 0; it < 4; ++it)
#pragma unroll
      for (int r = 0; r < 8; ++r) {
        float al = arow[it * 16 + r + hk * 8];
#pragma unroll
        for (int jt = 0; jt < 4; ++jt) acc[it][jt][r] *= al;
      }

#pragma unroll
    for (int ks = 0; ks < 2; ++ks) {
      v16bf pa[4];
#pragma unroll
      for (int it = 0; it < 4; ++it)
        pa[it] = load_frag_a(Psh + ks * 32, it * 16, LDP, lane);
#pragma unroll
      for (int jt = 0; jt < 4; ++jt) {
        v16bf bf = load_frag_b(vb + (size_t)(ch0 + jt * 16) * N_PIX + kt0 + ks * 32,
                               N_PIX, lane);
#pragma unroll
        for (int it = 0; it < 4; ++it)
          acc[it][jt] = WMMA_BF16(pa[it], bf, acc[it][jt]);
      }
    }
    cur ^= 1;
    // next iteration's barriers protect Ksh/Psh/arow reuse
  }

  // ---- finalize: out = gamma * O / l + x  (float2 pairs, 32-bit offsets) -
  const float* xb = x   + (size_t)b * CHN * N_PIX;
  float*       ob = out + (size_t)b * CHN * N_PIX;
#pragma unroll
  for (int it = 0; it < 4; ++it)
#pragma unroll
    for (int r = 0; r < 8; r += 2) {
      int   pix = q0 + it * 16 + r + hk * 8;
      float l0  = 1.f / lrow[it * 16 + r + hk * 8];
      float l1  = 1.f / lrow[it * 16 + r + 1 + hk * 8];
#pragma unroll
      for (int jt = 0; jt < 4; ++jt) {
        unsigned idx = (unsigned)((ch0 + jt * 16 + nn) * N_PIX + pix);
        float2 xv = *(const float2*)(xb + idx);
        float2 ov;
        ov.x = gamma * (acc[it][jt][r]     * l0) + xv.x;
        ov.y = gamma * (acc[it][jt][r + 1] * l1) + xv.y;
        *(float2*)(ob + idx) = ov;
      }
    }
}

// ---------------------------------------------------------------------------
extern "C" void kernel_launch(void* const* d_in, const int* in_sizes, int n_in,
                              void* d_out, int out_size, void* d_ws, size_t ws_size,
                              hipStream_t stream) {
  const float* x  = (const float*)d_in[0];
  const float* Wq = (const float*)d_in[1];
  const float* bq = (const float*)d_in[2];
  const float* Wk = (const float*)d_in[3];
  const float* bk = (const float*)d_in[4];
  const float* Wv = (const float*)d_in[5];
  const float* bv = (const float*)d_in[6];
  const float* gm = (const float*)d_in[7];
  float* out = (float*)d_out;

  // workspace layout (bf16): qt 2MB | kt 2MB | vt 16MB  (20MB total)
  bf16_t* qt = (bf16_t*)d_ws;
  bf16_t* kt = qt + (size_t)4 * N_PIX * 64;
  bf16_t* vt = kt + (size_t)4 * N_PIX * 64;

  dim3 pb(256), pg(N_PIX / 128, 10, 4);
  proj_kernel<<<pg, pb, 0, stream>>>(x, Wq, bq, Wk, bk, Wv, bv, qt, kt, vt);

  dim3 ab(256), ag(N_PIX / 64, 4);
  attn_kernel<<<ag, ab, 0, stream>>>(x, gm, qt, kt, vt, out);
}
